// MSAStack_15496242004725
// MI455X (gfx1250) — matrix operand order repaired and checked by reading, not compiled, not run on hardware
//
#include <hip/hip_runtime.h>
#include <hip/hip_bf16.h>

// ---------------------------------------------------------------------------
// MSAPairWeightedAveraging + SwiGLU transition for MI455X (gfx1250, wave32).
// All GEMMs via v_wmma_f32_16x16x32_bf16. Every WMMA operand is stored in
// fragment-contiguous ("swizzled") layout so each fragment is one 32-byte
// contiguous load per lane (2x b128). Epilogue scatters exploit tile
// alignment: per-lane offsets are base + r*32B (immediate offsets), and the
// K3 v-scatter collapses to a single 16-byte packed store per lane/tile.
//
// Fragment layouts (ISA 7.12.2, wave32):
//  A 16x32 bf16: lane = m + 16*hi, element e; k = 8*hi + e + 8*(e>=8)
//     inverse: hi = (k>>3)&1, e = (k&7) + 8*(k>>4)
//  B 32x16 bf16: lane = n + 16*hi, element e; k = 8*hi + e + 8*(e>=8)
//     inverse: hi = (k>>3)&1, e = (k&7) + 8*(k>>4)
//  C/D 16x16 f32: VGPR r: m = r + 8*(lane>=16), n = lane&15
// ---------------------------------------------------------------------------

typedef __bf16 bf16_t;
typedef __attribute__((ext_vector_type(16))) __bf16 v16bf;
typedef __attribute__((ext_vector_type(8)))  __bf16 v8bf;
typedef __attribute__((ext_vector_type(8)))  float  v8f;

#define S_DIM 1024
#define N_DIM 512
#define CM    64
#define CZ    128
#define HH    8
#define CC    8
#define HC    64     // H*C
#define HID   256
#define SC    8192   // S*C

__device__ __forceinline__ v8f wmma_bf16(v16bf a, v16bf b, v8f c) {
  return __builtin_amdgcn_wmma_f32_16x16x32_bf16(
      false, a, false, b, (short)0, c, false, false);
}

__device__ __forceinline__ v16bf frag_ld(const bf16_t* base, size_t frag,
                                         int lane) {
  return *(const v16bf*)(base + (frag * 32 + (size_t)lane) * 16);
}

__device__ __forceinline__ void a_inv(int k5, int& hi, int& e) {
  hi = (k5 >> 3) & 1;
  e  = (k5 & 7) + 8 * (k5 >> 4);
}
__device__ __forceinline__ void b_inv(int k5, int& hi, int& e) {
  hi = (k5 >> 3) & 1;
  e  = (k5 & 7) + 8 * (k5 >> 4);
}

// ---------------------------------------------------------------------------
__global__ __launch_bounds__(256) void k_wswz(
    const float* __restrict__ src, bf16_t* __restrict__ dst, int K, int N) {
  int idx = blockIdx.x * blockDim.x + threadIdx.x;
  if (idx >= K * N) return;
  int k = idx / N, n = idx % N;
  int hi, e; b_inv(k & 31, hi, e);
  int ntiles = N >> 4;
  dst[(((size_t)(k >> 5) * ntiles + (n >> 4)) * 32 + (n & 15) + 16 * hi) * 16 +
      e] = (bf16_t)src[idx];
}

// ---------------------------------------------------------------------------
__global__ __launch_bounds__(256) void k_bias(
    const float* __restrict__ z, const float* __restrict__ Wz,
    const float* __restrict__ lnzg, const float* __restrict__ lnzb,
    float* __restrict__ bias) {
  int wave = blockIdx.x * (blockDim.x >> 5) + (threadIdx.x >> 5);
  int lane = threadIdx.x & 31;
  int i = wave / N_DIM, j = wave % N_DIM;
  const float* zrow = z + ((size_t)i * N_DIM + j) * CZ;
  float x[4], s = 0.f, s2 = 0.f;
#pragma unroll
  for (int t = 0; t < 4; ++t) {
    x[t] = zrow[lane + 32 * t];
    s += x[t]; s2 += x[t] * x[t];
  }
#pragma unroll
  for (int msk = 16; msk >= 1; msk >>= 1) {
    s  += __shfl_xor(s, msk, 32);
    s2 += __shfl_xor(s2, msk, 32);
  }
  float mu   = s * (1.f / 128.f);
  float var  = s2 * (1.f / 128.f) - mu * mu;
  float rstd = rsqrtf(var + 1e-5f);
  float nz[4];
#pragma unroll
  for (int t = 0; t < 4; ++t) {
    int c = lane + 32 * t;
    nz[t] = (x[t] - mu) * rstd * lnzg[c] + lnzb[c];
  }
#pragma unroll
  for (int h = 0; h < HH; ++h) {
    float d = 0.f;
#pragma unroll
    for (int t = 0; t < 4; ++t) d += nz[t] * Wz[(lane + 32 * t) * HH + h];
#pragma unroll
    for (int msk = 16; msk >= 1; msk >>= 1) d += __shfl_xor(d, msk, 32);
    if (lane == 0) bias[((size_t)i * N_DIM + j) * HH + h] = d;
  }
}

// ---------------------------------------------------------------------------
__global__ __launch_bounds__(256) void k_softmax(
    const float* __restrict__ bias, bf16_t* __restrict__ w_sw) {
  int wave = blockIdx.x * (blockDim.x >> 5) + (threadIdx.x >> 5);
  int lane = threadIdx.x & 31;
  int i = wave >> 3, h = wave & 7;
  float v[16];
  float mx = -3.0e38f;
#pragma unroll
  for (int t = 0; t < 16; ++t) {
    int j = lane + 32 * t;
    v[t] = bias[((size_t)i * N_DIM + j) * HH + h];
    mx = fmaxf(mx, v[t]);
  }
#pragma unroll
  for (int msk = 16; msk >= 1; msk >>= 1) mx = fmaxf(mx, __shfl_xor(mx, msk, 32));
  float s = 0.f;
#pragma unroll
  for (int t = 0; t < 16; ++t) { v[t] = __expf(v[t] - mx); s += v[t]; }
#pragma unroll
  for (int msk = 16; msk >= 1; msk >>= 1) s += __shfl_xor(s, msk, 32);
  float inv = __fdividef(1.f, s);
  int it = i >> 4, mrow = i & 15;
  int hi, e; a_inv(lane, hi, e);
  bf16_t* p = w_sw + (((size_t)(h * 32 + it) * 16) * 32 + mrow + 16 * hi) * 16 + e;
#pragma unroll
  for (int t = 0; t < 16; ++t) p[(size_t)t * 512] = (bf16_t)(v[t] * inv);
}

// ---------------------------------------------------------------------------
__device__ __forceinline__ void ln_tile_swz(
    const float* __restrict__ rows_base, float* stage, bf16_t* asw,
    const float* __restrict__ gamma, const float* __restrict__ beta,
    int lane) {
#pragma unroll
  for (int t = 0; t < 32; ++t) stage[lane + 32 * t] = rows_base[lane + 32 * t];
  int r = lane & 15, half = (lane >> 4) & 1;
  float s = 0.f, s2 = 0.f;
#pragma unroll
  for (int t = 0; t < 32; ++t) {
    float x = stage[r * 64 + half * 32 + t];
    s += x; s2 += x * x;
  }
  s  += __shfl_xor(s, 16, 32);
  s2 += __shfl_xor(s2, 16, 32);
  float mu   = s * (1.f / 64.f);
  float var  = s2 * (1.f / 64.f) - mu * mu;
  float rstd = rsqrtf(var + 1e-5f);
#pragma unroll
  for (int t = 0; t < 32; ++t) {
    int c = half * 32 + t;
    float x = (stage[r * 64 + c] - mu) * rstd * gamma[c] + beta[c];
    int hi, e; a_inv(t, hi, e);
    asw[((size_t)half * 32 + r + 16 * hi) * 16 + e] = (bf16_t)x;
  }
}

// ---------------------------------------------------------------------------
__global__ __launch_bounds__(256) void k_mnvg(
    const float* __restrict__ m, const float* __restrict__ lnmg,
    const float* __restrict__ lnmb, const bf16_t* __restrict__ Wmv_sw,
    const bf16_t* __restrict__ Wmg_sw, bf16_t* __restrict__ v_sw,
    bf16_t* __restrict__ g_sw) {
  __shared__ __align__(32) float  stage_s[8][16 * 64];
  __shared__ __align__(32) bf16_t asw_s[8][16 * 64];
  int lane = threadIdx.x & 31, wv = threadIdx.x >> 5;
  int rowbase = (blockIdx.x * 8 + wv) * 16;
  ln_tile_swz(m + (size_t)rowbase * CM, stage_s[wv], asw_s[wv], lnmg, lnmb,
              lane);
  v16bf a0 = frag_ld(asw_s[wv], 0, lane);
  v16bf a1 = frag_ld(asw_s[wv], 1, lane);

  int hl = (lane >> 4) & 1;
  int s_idx = rowbase >> 9;
  int jn_b  = rowbase & (N_DIM - 1);
  int jn5   = jn_b >> 5;
  int hi_b  = (jn_b >> 4) & 1;
  size_t gfid0 = ((size_t)(rowbase >> 4)) * 2;

#pragma unroll
  for (int nt = 0; nt < 4; ++nt) {
    v8f accv = {};
    accv = wmma_bf16(a0, frag_ld(Wmv_sw, 0 * 4 + nt, lane), accv);
    accv = wmma_bf16(a1, frag_ld(Wmv_sw, 1 * 4 + nt, lane), accv);
    v8f accg = {};
    accg = wmma_bf16(a0, frag_ld(Wmg_sw, 0 * 4 + nt, lane), accg);
    accg = wmma_bf16(a1, frag_ld(Wmg_sw, 1 * 4 + nt, lane), accg);
    int col = nt * 16 + (lane & 15);
    int h = col >> 3, c = col & 7;
    int col_sc = (s_idx << 3) + c;
    size_t vfid = (size_t)h * (16 * (SC / 16)) + (size_t)jn5 * (SC / 16) +
                  (col_sc >> 4);
    v8bf vv;
#pragma unroll
    for (int r = 0; r < 8; ++r) vv[r] = (bf16_t)accv[r];
    *(v8bf*)(v_sw + (vfid * 32 + (col_sc & 15) + 16 * hl) * 16 + 8 * hi_b) = vv;
    int hi_a = (col >> 3) & 1;
    int e_a  = c + 8 * ((col >> 4) & 1);
    size_t goff = ((gfid0 + (col >> 5)) * 32 + 8 * hl + 16 * hi_a) * 16 + e_a;
#pragma unroll
    for (int r = 0; r < 8; ++r)
      g_sw[goff + (size_t)r * 16] =
          (bf16_t)__fdividef(1.f, 1.f + __expf(-accg[r]));
  }
}

// ---------------------------------------------------------------------------
__global__ __launch_bounds__(256) void k_einsum(
    const bf16_t* __restrict__ w_sw, const bf16_t* __restrict__ v_sw,
    const bf16_t* __restrict__ g_sw, bf16_t* __restrict__ o_sw) {
  int b = blockIdx.x;
  int cb = b & 15;
  int it = (b >> 4) & 31;
  int h  = b >> 9;
  int lane = threadIdx.x & 31, wv = threadIdx.x >> 5;

  const bf16_t* a_base = w_sw + ((size_t)(h * 32 + it) * 16) * 32 * 16;
  const bf16_t* b_base = v_sw + (size_t)h * 16 * (SC / 16) * 32 * 16;
  int nt0 = cb * 32 + wv * 4;

  v8f acc[4] = {};
  for (int kt = 0; kt < 16; ++kt) {
    v16bf a = frag_ld(a_base, kt, lane);
    __builtin_prefetch(
        (const void*)(b_base + (((size_t)(kt + 1) * (SC / 16) + nt0) * 32) * 16),
        0, 1);
#pragma unroll
    for (int u = 0; u < 4; ++u) {
      v16bf bf = frag_ld(b_base, (size_t)kt * (SC / 16) + nt0 + u, lane);
      acc[u] = wmma_bf16(a, bf, acc[u]);
    }
  }
  int hl = (lane >> 4) & 1;
  int hi_a = h & 1;
  int kt_o = h >> 2;
  int e_b  = 8 * ((h >> 1) & 1);
#pragma unroll
  for (int u = 0; u < 4; ++u) {
    int col = (nt0 + u) * 16 + (lane & 15);
    int s = col >> 3, c = col & 7;
    size_t off = ((((size_t)s * 32 + it) * 2 + kt_o) * 32 + 8 * hl +
                  16 * hi_a) * 16 + c + e_b;
    const bf16_t* gp = g_sw + off;
    bf16_t* op = o_sw + off;
#pragma unroll
    for (int r = 0; r < 8; ++r) {
      float gv = (float)gp[(size_t)r * 16];
      op[(size_t)r * 16] = (bf16_t)(acc[u][r] * gv);
    }
  }
}

// ---------------------------------------------------------------------------
__global__ __launch_bounds__(256) void k_wout(
    const bf16_t* __restrict__ o_sw, const bf16_t* __restrict__ Wout_sw,
    const float* __restrict__ m_in, float* __restrict__ out) {
  int lane = threadIdx.x & 31, wv = threadIdx.x >> 5;
  int rowbase = (blockIdx.x * 8 + wv) * 16;
  int hl = (lane >> 4) & 1;
  size_t rt = (size_t)(rowbase >> 4);
  v16bf a0 = frag_ld(o_sw, rt * 2 + 0, lane);
  v16bf a1 = frag_ld(o_sw, rt * 2 + 1, lane);
#pragma unroll
  for (int nt = 0; nt < 4; ++nt) {
    v8f acc = {};
    acc = wmma_bf16(a0, frag_ld(Wout_sw, 0 * 4 + nt, lane), acc);
    acc = wmma_bf16(a1, frag_ld(Wout_sw, 1 * 4 + nt, lane), acc);
    int n = nt * 16 + (lane & 15);
    const float* mp = m_in + (size_t)(rowbase + 8 * hl) * CM + n;
    float* op = out + (size_t)(rowbase + 8 * hl) * CM + n;
#pragma unroll
    for (int r = 0; r < 8; ++r)
      op[(size_t)r * CM] = mp[(size_t)r * CM] + acc[r];
  }
}

// ---------------------------------------------------------------------------
__global__ __launch_bounds__(128) void k_transition(
    float* __restrict__ out, const float* __restrict__ lntg,
    const float* __restrict__ lntb, const bf16_t* __restrict__ Wa_sw,
    const bf16_t* __restrict__ Wb_sw, const bf16_t* __restrict__ Wo_sw) {
  __shared__ __align__(32) char smem[4][10240];
  int lane = threadIdx.x & 31, wv = threadIdx.x >> 5;
  int rowbase = (blockIdx.x * 4 + wv) * 16;
  float*  stage = (float*)smem[wv];
  bf16_t* hid   = (bf16_t*)smem[wv];
  bf16_t* asw   = (bf16_t*)(smem[wv] + 8192);

  ln_tile_swz(out + (size_t)rowbase * CM, stage, asw, lntg, lntb, lane);
  v16bf a0 = frag_ld(asw, 0, lane);
  v16bf a1 = frag_ld(asw, 1, lane);

  int hl = (lane >> 4) & 1;
#pragma unroll 2
  for (int nt = 0; nt < 16; ++nt) {
    v8f aa = {};
    aa = wmma_bf16(a0, frag_ld(Wa_sw, 0 * 16 + nt, lane), aa);
    aa = wmma_bf16(a1, frag_ld(Wa_sw, 1 * 16 + nt, lane), aa);
    v8f bb = {};
    bb = wmma_bf16(a0, frag_ld(Wb_sw, 0 * 16 + nt, lane), bb);
    bb = wmma_bf16(a1, frag_ld(Wb_sw, 1 * 16 + nt, lane), bb);
    int col = nt * 16 + (lane & 15);
    int hi, e; a_inv(col & 31, hi, e);
    bf16_t* hp = hid + (((size_t)(col >> 5) * 32) + 8 * hl + 16 * hi) * 16 + e;
#pragma unroll
    for (int r = 0; r < 8; ++r) {
      float x = aa[r];
      hp[(size_t)r * 16] = (bf16_t)(__fdividef(x, 1.f + __expf(-x)) * bb[r]);
    }
  }

#pragma unroll
  for (int nt = 0; nt < 4; ++nt) {
    v8f acc = {};
#pragma unroll
    for (int kc = 0; kc < 8; ++kc) {
      v16bf a = frag_ld(hid, kc, lane);
      acc = wmma_bf16(a, frag_ld(Wo_sw, (size_t)kc * 4 + nt, lane), acc);
    }
    int n = nt * 16 + (lane & 15);
    float* op = out + (size_t)(rowbase + 8 * hl) * CM + n;
#pragma unroll
    for (int r = 0; r < 8; ++r)
      op[(size_t)r * CM] = op[(size_t)r * CM] + acc[r];
  }
}

// ---------------------------------------------------------------------------
extern "C" void kernel_launch(void* const* d_in, const int* in_sizes, int n_in,
                              void* d_out, int out_size, void* d_ws,
                              size_t ws_size, hipStream_t stream) {
  const float* m    = (const float*)d_in[0];
  const float* z    = (const float*)d_in[1];
  const float* lnmg = (const float*)d_in[2];
  const float* lnmb = (const float*)d_in[3];
  const float* Wmv  = (const float*)d_in[4];
  const float* lnzg = (const float*)d_in[5];
  const float* lnzb = (const float*)d_in[6];
  const float* Wz   = (const float*)d_in[7];
  const float* Wmg  = (const float*)d_in[8];
  const float* Wout = (const float*)d_in[9];
  const float* lntg = (const float*)d_in[10];
  const float* lntb = (const float*)d_in[11];
  const float* Wa   = (const float*)d_in[12];
  const float* Wb   = (const float*)d_in[13];
  const float* Wo   = (const float*)d_in[14];
  float* out = (float*)d_out;

  char* ws = (char*)d_ws;
  size_t off = 0;
  auto alloc = [&](size_t bytes) {
    void* p = ws + off;
    off = (off + bytes + 255) & ~(size_t)255;
    return p;
  };
  bf16_t* Wmv_sw  = (bf16_t*)alloc(CM * HC * 2);
  bf16_t* Wmg_sw  = (bf16_t*)alloc(CM * HC * 2);
  bf16_t* Wout_sw = (bf16_t*)alloc(HC * CM * 2);
  bf16_t* Wa_sw   = (bf16_t*)alloc(CM * HID * 2);
  bf16_t* Wb_sw   = (bf16_t*)alloc(CM * HID * 2);
  bf16_t* Wo_sw   = (bf16_t*)alloc(HID * CM * 2);
  float*  bias    = (float*)alloc((size_t)N_DIM * N_DIM * HH * 4);
  bf16_t* w_sw    = (bf16_t*)alloc((size_t)HH * N_DIM * N_DIM * 2);
  bf16_t* v_sw    = (bf16_t*)alloc((size_t)HH * N_DIM * SC * 2);
  bf16_t* g_sw    = (bf16_t*)alloc((size_t)HH * N_DIM * SC * 2);
  bf16_t* o_sw    = (bf16_t*)alloc((size_t)S_DIM * N_DIM * HC * 2);

  auto swz = [&](const float* s, bf16_t* d, int K, int N) {
    k_wswz<<<(K * N + 255) / 256, 256, 0, stream>>>(s, d, K, N);
  };
  swz(Wmv, Wmv_sw, CM, HC);
  swz(Wmg, Wmg_sw, CM, HC);
  swz(Wout, Wout_sw, HC, CM);
  swz(Wa, Wa_sw, CM, HID);
  swz(Wb, Wb_sw, CM, HID);
  swz(Wo, Wo_sw, HID, CM);

  k_bias<<<(N_DIM * N_DIM) / 8, 256, 0, stream>>>(z, Wz, lnzg, lnzb, bias);
  k_softmax<<<(N_DIM * HH) / 8, 256, 0, stream>>>(bias, w_sw);
  k_mnvg<<<(S_DIM * N_DIM) / 128, 256, 0, stream>>>(m, lnmg, lnmb, Wmv_sw,
                                                    Wmg_sw, v_sw, g_sw);
  k_einsum<<<HH * 32 * 16, 256, 0, stream>>>(w_sw, v_sw, g_sw, o_sw);
  k_wout<<<(S_DIM * N_DIM) / 128, 256, 0, stream>>>(o_sw, Wout_sw, m, out);
  k_transition<<<(S_DIM * N_DIM) / 64, 128, 0, stream>>>(out, lntg, lntb,
                                                         Wa_sw, Wb_sw, Wo_sw);
}
